// Net_68289980006701
// MI455X (gfx1250) — matrix operand order repaired and checked by reading, not compiled, not run on hardware
//
#include <hip/hip_runtime.h>
#include <hip/hip_bf16.h>

typedef __attribute__((ext_vector_type(16))) _Float16 v16h;
typedef __attribute__((ext_vector_type(8)))  _Float16 v8h;
typedef __attribute__((ext_vector_type(8)))  float    v8f;

#define WAVES 4
#define TPB   (WAVES * 32)

struct Acc64 { v8f t[4]; };

// Branch-free activations. Prefer the gfx1250 native v_tanh_f32 (TRANS32 pipe,
// co-executes with WMMA); fall back to exp/rcp form if the builtin is absent.
__device__ __forceinline__ float fast_tanh(float x) {
#if __has_builtin(__builtin_amdgcn_tanhf)
  return __builtin_amdgcn_tanhf(x);
#else
  float t = __expf(2.f * x);
  return 1.f - 2.f * __builtin_amdgcn_rcpf(t + 1.f);
#endif
}
__device__ __forceinline__ float fast_sigmoid(float x) {
  return __builtin_amdgcn_rcpf(1.f + __expf(-x));
}

// ---- WMMA fragment loaders (CDNA5 wave32 layouts, CDNA5 ISA) ----

// 16-bit A matrix 16x32: lanes 0-15 hold M=lane, K=0..7 (VGPR0-3) and K=16..23
// (VGPR4-7); lanes 16-31 hold K=8..15 and K=24..31. Source is row-major
// [16 rows x ldk halfs] in LDS => two aligned 16B loads per lane.
__device__ __forceinline__ v16h ldsA(const _Float16* m, int ldk, int kbase, int lane) {
  int row = lane & 15;
  int k0  = kbase + ((lane & 16) ? 8 : 0);
  const _Float16* p = m + row * ldk + k0;
  v8h lo = *(const v8h*)(p);
  v8h hi = *(const v8h*)(p + 16);
  v16h a;
#pragma unroll
  for (int i = 0; i < 8; ++i) { a[i] = lo[i]; a[i + 8] = hi[i]; }
  return a;
}

// 16-bit B matrix 32x16: lanes 0-15 hold N=lane, K=0..7 and K=16..23;
// lanes 16-31 hold K=8..15 and K=24..31. Weights stored transposed [N][ldk] in LDS.
__device__ __forceinline__ v16h ldsB(const _Float16* wt, int ldk, int kbase, int nbase, int lane) {
  int n  = nbase + (lane & 15);
  int k0 = kbase + ((lane & 16) ? 8 : 0);
  const _Float16* p = wt + n * ldk + k0;
  v8h lo = *(const v8h*)(p);
  v8h hi = *(const v8h*)(p + 16);
  v16h b;
#pragma unroll
  for (int i = 0; i < 8; ++i) { b[i] = lo[i]; b[i + 8] = hi[i]; }
  return b;
}

// out = tanh(X(16xK=32, zero padded) @ W(32x64) + bias), C layout.
__device__ __forceinline__ void layer_k32(const _Float16* x, const _Float16* wt,
                                          const float* bias, int lane, Acc64& h) {
  v16h a = ldsA(x, 32, 0, lane);
#pragma unroll
  for (int nt = 0; nt < 4; ++nt) {
    v16h b = ldsB(wt, 32, 0, nt * 16, lane);
    v8f c = {};
    c = __builtin_amdgcn_wmma_f32_16x16x32_f16(false, a, false, b, (short)0, c, false, false);
    float bb = bias[nt * 16 + (lane & 15)];
#pragma unroll
    for (int j = 0; j < 8; ++j) h.t[nt][j] = fast_tanh(c[j] + bb);
  }
}

// out = tanh(X(16x64) @ W(64x64) + bias), two K=32 WMMA accumulation steps.
__device__ __forceinline__ void layer_k64(const _Float16* x, const _Float16* wt,
                                          const float* bias, int lane, Acc64& h) {
  v16h a0 = ldsA(x, 64, 0,  lane);
  v16h a1 = ldsA(x, 64, 32, lane);
#pragma unroll
  for (int nt = 0; nt < 4; ++nt) {
    v16h b0 = ldsB(wt, 64, 0,  nt * 16, lane);
    v16h b1 = ldsB(wt, 64, 32, nt * 16, lane);
    v8f c = {};
    c = __builtin_amdgcn_wmma_f32_16x16x32_f16(false, a0, false, b0, (short)0, c, false, false);
    c = __builtin_amdgcn_wmma_f32_16x16x32_f16(false, a1, false, b1, (short)0, c, false, false);
    float bb = bias[nt * 16 + (lane & 15)];
#pragma unroll
    for (int j = 0; j < 8; ++j) h.t[nt][j] = fast_tanh(c[j] + bb);
  }
}

// Spill C-layout 16x64 activation tile to row-major f16 LDS (the M<->lane transpose).
__device__ __forceinline__ void stageH(_Float16* x, const Acc64& h, int lane) {
  int col = lane & 15;
  int mb  = (lane & 16) ? 8 : 0;
#pragma unroll
  for (int nt = 0; nt < 4; ++nt)
#pragma unroll
    for (int j = 0; j < 8; ++j)
      x[(mb + j) * 64 + nt * 16 + col] = (_Float16)h.t[nt][j];
}

// Fused head: [W2a | W2b | 0] as 64x16 tile -> one 16x16 C tile. Col 0..7 = v, col 8 = w-logit.
__device__ __forceinline__ v8f head(const _Float16* x, const _Float16* wt2,
                                    const float* b2, int lane) {
  v16h a0 = ldsA(x, 64, 0,  lane);
  v16h a1 = ldsA(x, 64, 32, lane);
  v16h b0 = ldsB(wt2, 64, 0,  0, lane);
  v16h b1 = ldsB(wt2, 64, 32, 0, lane);
  v8f c = {};
  c = __builtin_amdgcn_wmma_f32_16x16x32_f16(false, a0, false, b0, (short)0, c, false, false);
  c = __builtin_amdgcn_wmma_f32_16x16x32_f16(false, a1, false, b1, (short)0, c, false, false);
  float bb = b2[lane & 15];
#pragma unroll
  for (int j = 0; j < 8; ++j) c[j] += bb;
  return c;
}

// Transposed (output-major) f16 weight fill with K zero padding: dst[n][k] = src[k][n].
__device__ __forceinline__ void fill_wt(_Float16* dst, const float* src, int K, int N, int Kpad) {
  for (int i = threadIdx.x; i < N * Kpad; i += blockDim.x) {
    int n = i / Kpad, k = i - n * Kpad;
    dst[i] = (k < K) ? (_Float16)src[k * N + n] : (_Float16)0.f;
  }
}
__device__ __forceinline__ void fill_b(float* dst, const float* src, int n) {
  for (int i = threadIdx.x; i < n; i += blockDim.x) dst[i] = src[i];
}

__global__ __launch_bounds__(TPB) void net_wmma_kernel(
    const float* __restrict__ states, const float* __restrict__ action,
    const float* __restrict__ b_type, const float* __restrict__ table,
    const float* __restrict__ W1,  const float* __restrict__ b1,
    const float* __restrict__ W1b, const float* __restrict__ b1b,
    const float* __restrict__ W1c, const float* __restrict__ b1c,
    const float* __restrict__ W2a, const float* __restrict__ b2a,
    const float* __restrict__ W2b, const float* __restrict__ b2b,
    const float* __restrict__ Ta1, const float* __restrict__ ta1,
    const float* __restrict__ Ta2, const float* __restrict__ ta2,
    const float* __restrict__ Tb1, const float* __restrict__ tb1,
    const float* __restrict__ Tb2, const float* __restrict__ tb2,
    float* __restrict__ out, int Bn) {
  __shared__ __align__(16) _Float16 sW1[64 * 32], sW1b[64 * 64], sW1c[64 * 64];
  __shared__ __align__(16) _Float16 sTa1[64 * 32], sTa2[64 * 64];
  __shared__ __align__(16) _Float16 sTb1[64 * 32], sTb2[64 * 64];
  __shared__ __align__(16) _Float16 sW2[16 * 64];
  __shared__ float sb1[64], sb1b[64], sb1c[64], sta1[64], sta2[64], stb1[64], stb2[64], sb2[16];
  __shared__ __align__(16) _Float16 sX1[WAVES * 16 * 32];  // first-layer A staging (K padded to 32)
  __shared__ __align__(16) _Float16 sX2[WAVES * 16 * 64];  // hidden-layer A staging (K = 64)

  // ---- block-cooperative weight prep (f32 -> f16, transposed, zero-padded) ----
  fill_wt(sW1,  W1,  14, 64, 32);
  fill_wt(sW1b, W1b, 64, 64, 64);
  fill_wt(sW1c, W1c, 64, 64, 64);
  fill_wt(sTa1, Ta1,  5, 64, 32);
  fill_wt(sTa2, Ta2, 64, 64, 64);
  fill_wt(sTb1, Tb1,  5, 64, 32);
  fill_wt(sTb2, Tb2, 64, 64, 64);
  for (int i = threadIdx.x; i < 16 * 64; i += blockDim.x) {  // fused head [W2a|W2b|0]
    int n = i / 64, k = i - n * 64;
    float v = (n < 8) ? W2a[k * 8 + n] : ((n == 8) ? W2b[k] : 0.f);
    sW2[i] = (_Float16)v;
  }
  fill_b(sb1,  b1,  64); fill_b(sb1b, b1b, 64); fill_b(sb1c, b1c, 64);
  fill_b(sta1, ta1, 64); fill_b(sta2, ta2, 64);
  fill_b(stb1, tb1, 64); fill_b(stb2, tb2, 64);
  for (int i = threadIdx.x; i < 16; i += blockDim.x)
    sb2[i] = (i < 8) ? b2a[i] : ((i == 8) ? b2b[0] : 0.f);
  __syncthreads();

  const int lane = threadIdx.x & 31;
  const int wave = threadIdx.x >> 5;
  const int base = (blockIdx.x * WAVES + wave) * 16;  // 16 batch rows per wave
  _Float16* xb1 = sX1 + wave * 16 * 32;
  _Float16* xb2 = sX2 + wave * 16 * 64;

  // zero the K pad region of the first-layer staging once (same-wave LDS is in-order)
  for (int i = lane; i < 16 * 32; i += 32) xb1[i] = (_Float16)0.f;

  if (base >= Bn) return;  // uniform per wave: EXEC stays all-ones for WMMA

  __builtin_prefetch(states + (size_t)base * 54, 0, 1);
  __builtin_prefetch(action + (size_t)base * 8, 0, 1);

  Acc64 g1, tg, h;

  // ---- gate1 = tanh(tanh(b_type @ Ta1 + ta1) @ Ta2 + ta2) ----
  if (lane < 16 && base + lane < Bn) {
    const float* bt = b_type + (size_t)(base + lane) * 5;
    _Float16* row = xb1 + lane * 32;
#pragma unroll
    for (int k = 0; k < 5; ++k) row[k] = (_Float16)bt[k];
#pragma unroll
    for (int k = 5; k < 14; ++k) row[k] = (_Float16)0.f;
  }
  layer_k32(xb1, sTa1, sta1, lane, g1);
  stageH(xb2, g1, lane);
  layer_k64(xb2, sTa2, sta2, lane, g1);

  v8f res = {};

  // ---- h_t pass: x = [action(8), table(6)] ----
  if (lane < 16 && base + lane < Bn) {
    int b = base + lane;
    _Float16* row = xb1 + lane * 32;
#pragma unroll
    for (int k = 0; k < 8; ++k) row[k] = (_Float16)action[b * 8 + k];
#pragma unroll
    for (int k = 0; k < 6; ++k) row[8 + k] = (_Float16)table[b * 6 + k];
  }
  layer_k32(xb1, sW1, sb1, lane, h);
  stageH(xb2, h, lane);
  layer_k64(xb2, sW1b, sb1b, lane, h);
#pragma unroll
  for (int nt = 0; nt < 4; ++nt)
#pragma unroll
    for (int j = 0; j < 8; ++j) h.t[nt][j] *= g1.t[nt][j];
  stageH(xb2, h, lane);
  layer_k64(xb2, sW1c, sb1c, lane, h);
  stageH(xb2, h, lane);
  {
    v8f c = head(xb2, sW2, sb2, lane);
#pragma unroll
    for (int j = 0; j < 8; ++j) {
      float logit = __shfl(c[j], (lane & 16) + 8, 32);  // broadcast col 8 within row
      float w = fast_sigmoid(logit);
      res[j] += w * c[j];
    }
  }

  // ---- per-row passes r = 0..2 ----
  for (int r = 0; r < 3; ++r) {
    // type gate: tanh(tanh(types_r @ Tb1 + tb1) @ Tb2 + tb2)
    if (lane < 16 && base + lane < Bn) {
      const float* sp = states + (size_t)(base + lane) * 54 + r * 18;
      _Float16* row = xb1 + lane * 32;
#pragma unroll
      for (int k = 0; k < 5; ++k) row[k] = (_Float16)sp[13 + k];
#pragma unroll
      for (int k = 5; k < 14; ++k) row[k] = (_Float16)0.f;
    }
    layer_k32(xb1, sTb1, stb1, lane, tg);
    stageH(xb2, tg, lane);
    layer_k64(xb2, sTb2, stb2, lane, tg);

    // stack input: [a0-s0, a1-s1, a2..a7, st(6)]
    if (lane < 16 && base + lane < Bn) {
      int b = base + lane;
      const float* sp = states + (size_t)b * 54 + r * 18;
      const float* ap = action + (size_t)b * 8;
      _Float16* row = xb1 + lane * 32;
      row[0] = (_Float16)(ap[0] - sp[0]);
      row[1] = (_Float16)(ap[1] - sp[1]);
#pragma unroll
      for (int k = 2; k < 8; ++k) row[k] = (_Float16)ap[k];
#pragma unroll
      for (int k = 0; k < 6; ++k) row[8 + k] = (_Float16)sp[3 + k];
    }
    layer_k32(xb1, sW1, sb1, lane, h);
    stageH(xb2, h, lane);
    layer_k64(xb2, sW1b, sb1b, lane, h);
#pragma unroll
    for (int nt = 0; nt < 4; ++nt)
#pragma unroll
      for (int j = 0; j < 8; ++j) h.t[nt][j] *= g1.t[nt][j];
    stageH(xb2, h, lane);
    layer_k64(xb2, sW1c, sb1c, lane, h);
#pragma unroll
    for (int nt = 0; nt < 4; ++nt)
#pragma unroll
      for (int j = 0; j < 8; ++j) h.t[nt][j] *= tg.t[nt][j];
    stageH(xb2, h, lane);

    v8f c = head(xb2, sW2, sb2, lane);
#pragma unroll
    for (int j = 0; j < 8; ++j) {
      int m = j + ((lane & 16) ? 8 : 0);
      int b = base + m;
      float st0 = (b < Bn) ? states[(size_t)b * 54 + r * 18 + 3] : -1.f;
      float logit = __shfl(c[j], (lane & 16) + 8, 32);
      float w = (st0 != -1.f) ? fast_sigmoid(logit) : 0.f;
      res[j] += w * c[j];
    }
  }

  // ---- residuals + action -> out (lanes with col < 8 write their column) ----
  {
    int col = lane & 15;
    if (col < 8) {
      int mb = (lane & 16) ? 8 : 0;
#pragma unroll
      for (int j = 0; j < 8; ++j) {
        int b = base + mb + j;
        if (b < Bn) out[(size_t)b * 8 + col] = res[j] + action[(size_t)b * 8 + col];
      }
    }
  }
}

extern "C" void kernel_launch(void* const* d_in, const int* in_sizes, int n_in,
                              void* d_out, int out_size, void* d_ws, size_t ws_size,
                              hipStream_t stream) {
  (void)n_in; (void)out_size; (void)d_ws; (void)ws_size;
  const float* states = (const float*)d_in[0];
  const float* action = (const float*)d_in[1];
  const float* b_type = (const float*)d_in[2];
  const float* table  = (const float*)d_in[3];
  const float* W1  = (const float*)d_in[4];
  const float* b1  = (const float*)d_in[5];
  const float* W1b = (const float*)d_in[6];
  const float* b1b = (const float*)d_in[7];
  const float* W1c = (const float*)d_in[8];
  const float* b1c = (const float*)d_in[9];
  const float* W2a = (const float*)d_in[10];
  const float* b2a = (const float*)d_in[11];
  const float* W2b = (const float*)d_in[12];
  const float* b2b = (const float*)d_in[13];
  const float* Ta1 = (const float*)d_in[14];
  const float* ta1 = (const float*)d_in[15];
  const float* Ta2 = (const float*)d_in[16];
  const float* ta2 = (const float*)d_in[17];
  const float* Tb1 = (const float*)d_in[18];
  const float* tb1 = (const float*)d_in[19];
  const float* Tb2 = (const float*)d_in[20];
  const float* tb2 = (const float*)d_in[21];

  int Bn = in_sizes[1] / 8;                 // action is (B, 8)
  int tiles = (Bn + 15) / 16;               // 16 batch rows per wave
  int blocks = (tiles + WAVES - 1) / WAVES; // WAVES waves per block

  net_wmma_kernel<<<blocks, TPB, 0, stream>>>(
      states, action, b_type, table, W1, b1, W1b, b1b, W1c, b1c,
      W2a, b2a, W2b, b2b, Ta1, ta1, Ta2, ta2, Tb1, tb1, Tb2, tb2,
      (float*)d_out, Bn);
}
